// Net_68384469287506
// MI455X (gfx1250) — matrix operand order repaired and checked by reading, not compiled, not run on hardware
//
#include <hip/hip_runtime.h>

// ---------------- WMMA types / helpers ----------------
typedef __attribute__((ext_vector_type(16))) __bf16 v16bf;
typedef __attribute__((ext_vector_type(8)))  float  v8f;

// Native conversion: let clang lower to v_cvt_*bf16* on gfx1250 (RNE).
__device__ __forceinline__ __bf16 f2bf(float f) { return (__bf16)f; }

// ---------------- small utility kernels ----------------
__global__ void fill_kernel(float* p, float v, int n) {
  int t = blockIdx.x * blockDim.x + threadIdx.x;
  if (t < n) p[t] = v;
}

__global__ void deg_edge_kernel(const int* __restrict__ col, float* __restrict__ deg, int nE) {
  int t = blockIdx.x * blockDim.x + threadIdx.x;
  if (t < nE) atomicAdd(&deg[col[t]], 1.0f);
}

__global__ void rsqrt_kernel(float* deg, int n) {
  int t = blockIdx.x * blockDim.x + threadIdx.x;
  if (t < n) { float d = deg[t]; deg[t] = d > 0.f ? rsqrtf(d) : 0.f; }
}

// Pre-pack 64x64 f32 weight into the exact bf16 B-fragment layout:
// Wp[((ct*2+kh)*32 + lane)*16 + e] = bf16( W[kh*32 + (e&7) + (e>=8?16:0) + (lane>=16?8:0)][ct*16 + lane%16] )
__global__ void pack_w_kernel(const float* __restrict__ W, __bf16* __restrict__ Wp) {
  int t = blockIdx.x * blockDim.x + threadIdx.x;
  if (t >= 4096) return;
  int e = t & 15, lane = (t >> 4) & 31, kh = (t >> 9) & 1, ct = t >> 10;
  int k = (e & 7) + ((e >= 8) ? 16 : 0) + ((lane >= 16) ? 8 : 0) + kh * 32;
  int col = ct * 16 + (lane & 15);
  Wp[t] = f2bf(W[k * 64 + col]);
}

// ---------------- WMMA GEMM: Y[N,64] = dinv[:,None] * (X[N,64] @ W[64,64]) --------
// 8 waves / 256-thread block, one 16-row strip per wave.
// 4 col tiles x 2 K-steps = 8 v_wmma per wave, f32 accumulate, dinv-scaled epilogue.
__global__ __launch_bounds__(256)
void gemm_wmma_kernel(const float* __restrict__ X, const __bf16* __restrict__ Wp,
                      const float* __restrict__ dinv, float* __restrict__ Y, int N) {
  const int lane  = threadIdx.x & 31;
  const int strip = blockIdx.x * 8 + (threadIdx.x >> 5);
  const int rb    = strip * 16;
  if (rb >= N) return;                     // wave-uniform exit; EXEC all-1 below

  const int hi   = lane >> 4;              // half-wave select
  const int mrow = lane & 15;
  int r = rb + mrow; if (r > N - 1) r = N - 1;

  // A (16x64 bf16, two 16x32 fragments). Per 16-bit A layout, lane's elements
  // form contiguous 8-float runs -> float4 loads.
  const float4* x4 = (const float4*)(X + (size_t)r * 64);
  v16bf a[2];
  #pragma unroll
  for (int j = 0; j < 4; ++j) {            // run j covers K in [j*16 + hi*8, +8)
    float4 p0 = x4[j * 4 + hi * 2];
    float4 p1 = x4[j * 4 + hi * 2 + 1];
    int f = j >> 1, eb = (j & 1) * 8;
    a[f][eb + 0] = f2bf(p0.x); a[f][eb + 1] = f2bf(p0.y);
    a[f][eb + 2] = f2bf(p0.z); a[f][eb + 3] = f2bf(p0.w);
    a[f][eb + 4] = f2bf(p1.x); a[f][eb + 5] = f2bf(p1.y);
    a[f][eb + 6] = f2bf(p1.z); a[f][eb + 7] = f2bf(p1.w);
  }

  v8f acc[4];
  #pragma unroll
  for (int ct = 0; ct < 4; ++ct) {
    v16bf b0 = *(const v16bf*)(Wp + ((size_t)(ct * 2 + 0) * 32 + lane) * 16);
    v16bf b1 = *(const v16bf*)(Wp + ((size_t)(ct * 2 + 1) * 32 + lane) * 16);
    v8f c = {};
    c = __builtin_amdgcn_wmma_f32_16x16x32_bf16(false, a[0], false, b0, (short)0, c, false, false);
    c = __builtin_amdgcn_wmma_f32_16x16x32_bf16(false, a[1], false, b1, (short)0, c, false, false);
    acc[ct] = c;
  }

  // D layout: lane holds col ct*16+mrow; VGPR v holds row v + 8*hi.
  // Epilogue: scale each output row by dinv[row].
  #pragma unroll
  for (int v = 0; v < 8; ++v) {
    int row = rb + v + 8 * hi;
    if (row < N) {
      float dv = dinv[row];
      #pragma unroll
      for (int ct = 0; ct < 4; ++ct)
        Y[(size_t)row * 64 + ct * 16 + mrow] = dv * acc[ct][v];
    }
  }
}

// ---------------- GCN aggregation (Y is already dinv-scaled) ----------------
// self-loop: agg[n,:] = dinv[n] * Y[n,:]   (== dinv[n]^2 * xw[n,:])
__global__ void selfloop_kernel(const float* __restrict__ Y, const float* __restrict__ dinv,
                                float* __restrict__ agg, int n16) {
  int t = blockIdx.x * blockDim.x + threadIdx.x;
  if (t >= n16) return;                       // t indexes float4 chunks
  float dv = dinv[t >> 4];
  float4 v = ((const float4*)Y)[t];
  float4 o; o.x = dv * v.x; o.y = dv * v.y; o.z = dv * v.z; o.w = dv * v.w;
  ((float4*)agg)[t] = o;
}

// 16 threads/edge, 4 channels each: float4 gather + 4 global atomic f32 adds.
// message = dinv[col] * Y[row,:]  (== dinv[row]*dinv[col]*xw[row,:])
__global__ void edge_scatter_kernel(const int* __restrict__ ei, const float* __restrict__ dinv,
                                    const float* __restrict__ Y, float* __restrict__ agg, int nE) {
  int t = blockIdx.x * blockDim.x + threadIdx.x;
  int e = t >> 4;
  if (e >= nE) return;
  int c4 = (t & 15) << 2;
  int r = ei[e];
  int c = ei[nE + e];
  float s = dinv[c];
  float4 v = *(const float4*)(Y + (size_t)r * 64 + c4);
  float* o = agg + (size_t)c * 64 + c4;
  atomicAdd(o + 0, s * v.x);
  atomicAdd(o + 1, s * v.y);
  atomicAdd(o + 2, s * v.z);
  atomicAdd(o + 3, s * v.w);
}

// out[n,c] = relu(bn(agg[n,c] + bias[c]))  (vectorized: 16 float4 per row)
__global__ void bias_bn_relu_kernel(const float* __restrict__ agg, const float* __restrict__ bias,
                                    const float* __restrict__ g, const float* __restrict__ bb,
                                    const float* __restrict__ m, const float* __restrict__ v,
                                    float* __restrict__ out, int n16) {
  int t = blockIdx.x * blockDim.x + threadIdx.x;
  if (t >= n16) return;
  int cq = t & 15;                            // float4 index within row
  float4 a  = ((const float4*)agg)[t];
  float4 b4 = ((const float4*)bias)[cq];
  float4 g4 = ((const float4*)g)[cq];
  float4 bb4= ((const float4*)bb)[cq];
  float4 m4 = ((const float4*)m)[cq];
  float4 v4 = ((const float4*)v)[cq];
  float4 o;
  o.x = fmaxf((a.x + b4.x - m4.x) * rsqrtf(v4.x + 1e-5f) * g4.x + bb4.x, 0.f);
  o.y = fmaxf((a.y + b4.y - m4.y) * rsqrtf(v4.y + 1e-5f) * g4.y + bb4.y, 0.f);
  o.z = fmaxf((a.z + b4.z - m4.z) * rsqrtf(v4.z + 1e-5f) * g4.z + bb4.z, 0.f);
  o.w = fmaxf((a.w + b4.w - m4.w) * rsqrtf(v4.w + 1e-5f) * g4.w + bb4.w, 0.f);
  ((float4*)out)[t] = o;
}

// ---------------- pooling ----------------
__global__ void count_kernel(const int* __restrict__ batch, float* __restrict__ cnt, int N) {
  int t = blockIdx.x * blockDim.x + threadIdx.x;
  if (t < N) atomicAdd(&cnt[batch[t]], 1.0f);
}

__global__ void pool_sum_kernel(const int* __restrict__ batch, const float* __restrict__ h,
                                float* __restrict__ pooled, int n64) {
  int t = blockIdx.x * blockDim.x + threadIdx.x;
  if (t >= n64) return;
  int n = t >> 6, c = t & 63;
  atomicAdd(&pooled[(size_t)batch[n] * 64 + c], h[t]);
}

__global__ void pool_div_kernel(float* pooled, const float* __restrict__ cnt, int g64) {
  int t = blockIdx.x * blockDim.x + threadIdx.x;
  if (t >= g64) return;
  pooled[t] /= fmaxf(cnt[t >> 6], 1.0f);
}

// ---------------- MLP head: mode 0 = +BN+ReLU, 1 = +ReLU, 2 = plain ----------------
__global__ void head_linear_kernel(const float* __restrict__ in, const float* __restrict__ Wm,
                                   const float* __restrict__ bias,
                                   const float* g, const float* bb, const float* m, const float* vv,
                                   float* __restrict__ out, int G, int IN, int OUT, int mode) {
  int t = blockIdx.x * blockDim.x + threadIdx.x;
  if (t >= G * OUT) return;
  int gi = t / OUT, j = t - gi * OUT;
  float s = bias[j];
  const float* xi = in + (size_t)gi * IN;
  for (int k = 0; k < IN; ++k) s = fmaf(xi[k], Wm[(size_t)k * OUT + j], s);
  if (mode == 0) s = (s - m[j]) * rsqrtf(vv[j] + 1e-5f) * g[j] + bb[j];
  if (mode <= 1) s = fmaxf(s, 0.f);
  out[t] = s;
}

// ---------------- host launch ----------------
extern "C" void kernel_launch(void* const* d_in, const int* in_sizes, int n_in,
                              void* d_out, int out_size, void* d_ws, size_t ws_size,
                              hipStream_t stream) {
  (void)n_in; (void)ws_size;
  const float* x    = (const float*)d_in[0];
  const int*   ei   = (const int*)d_in[1];
  const int*   batch= (const int*)d_in[2];
  const float* W0   = (const float*)d_in[3];
  const float* b0   = (const float*)d_in[4];
  const float* g0   = (const float*)d_in[5];
  const float* bb0  = (const float*)d_in[6];
  const float* m0   = (const float*)d_in[7];
  const float* v0   = (const float*)d_in[8];
  const float* W1   = (const float*)d_in[9];
  const float* b1   = (const float*)d_in[10];
  const float* g1   = (const float*)d_in[11];
  const float* bb1  = (const float*)d_in[12];
  const float* m1   = (const float*)d_in[13];
  const float* v1   = (const float*)d_in[14];
  const float* hW1  = (const float*)d_in[15];
  const float* hb1  = (const float*)d_in[16];
  const float* hg1  = (const float*)d_in[17];
  const float* hbb1 = (const float*)d_in[18];
  const float* hm1  = (const float*)d_in[19];
  const float* hv1  = (const float*)d_in[20];
  const float* hW2  = (const float*)d_in[21];
  const float* hb2  = (const float*)d_in[22];
  const float* hg2  = (const float*)d_in[23];
  const float* hbb2 = (const float*)d_in[24];
  const float* hm2  = (const float*)d_in[25];
  const float* hv2  = (const float*)d_in[26];
  const float* hW3  = (const float*)d_in[27];
  const float* hb3  = (const float*)d_in[28];
  const float* hW4  = (const float*)d_in[29];
  const float* hb4  = (const float*)d_in[30];

  const int N   = in_sizes[0] / 64;   // 50000 nodes
  const int nE  = in_sizes[1] / 2;    // 800000 edges
  const int G   = out_size;           // 256 graphs
  const int n64 = N * 64;
  const int n16 = N * 16;             // float4 chunks
  const int g64 = G * 64;

  // workspace carve-up (floats)
  float* ws     = (float*)d_ws;
  float* dinv   = ws;                                  // N (also deg)
  float* A      = dinv + (size_t)((N + 15) & ~15);     // N*64
  float* B      = A + (size_t)n64;                     // N*64
  float* pooled = B + (size_t)n64;                     // G*64
  float* cnt    = pooled + g64;                        // G
  float* z1     = cnt + G;                             // G*256
  float* z2     = z1 + (size_t)G * 256;                // G*128
  float* z3     = z2 + (size_t)G * 128;                // G*64
  __bf16* Wp0   = (__bf16*)(z3 + g64);                 // 4096 bf16
  __bf16* Wp1   = Wp0 + 4096;                          // 4096 bf16

  const int BLK = 256;
  int gbN   = (N + BLK - 1) / BLK;
  int gbE   = (nE + BLK - 1) / BLK;
  int gbN64 = (n64 + BLK - 1) / BLK;
  int gbN16 = (n16 + BLK - 1) / BLK;
  int gbE16 = (nE * 16 + BLK - 1) / BLK;
  int gbGemm = ((N + 15) / 16 + 7) / 8;                // 8 strips (waves) per block

  // degrees (shared by both layers): deg = 1 (self loop) + in-degree, then rsqrt
  fill_kernel<<<gbN, BLK, 0, stream>>>(dinv, 1.0f, N);
  deg_edge_kernel<<<gbE, BLK, 0, stream>>>(ei + nE, dinv, nE);
  rsqrt_kernel<<<gbN, BLK, 0, stream>>>(dinv, N);

  // pre-pack weights into bf16 WMMA B-fragment layout
  pack_w_kernel<<<16, BLK, 0, stream>>>(W0, Wp0);
  pack_w_kernel<<<16, BLK, 0, stream>>>(W1, Wp1);

  // ---- GCN layer 0 ----
  gemm_wmma_kernel<<<gbGemm, BLK, 0, stream>>>(x, Wp0, dinv, A, N);        // A = dinv .* (x@W0)
  selfloop_kernel<<<gbN16, BLK, 0, stream>>>(A, dinv, B, n16);             // B = dinv .* A
  edge_scatter_kernel<<<gbE16, BLK, 0, stream>>>(ei, dinv, A, B, nE);      // B[c] += dinv[c]*A[r]
  bias_bn_relu_kernel<<<gbN16, BLK, 0, stream>>>(B, b0, g0, bb0, m0, v0, A, n16);  // A = h0

  // ---- GCN layer 1 ----
  gemm_wmma_kernel<<<gbGemm, BLK, 0, stream>>>(A, Wp1, dinv, B, N);        // B = dinv .* (h0@W1)
  selfloop_kernel<<<gbN16, BLK, 0, stream>>>(B, dinv, A, n16);
  edge_scatter_kernel<<<gbE16, BLK, 0, stream>>>(ei, dinv, B, A, nE);
  bias_bn_relu_kernel<<<gbN16, BLK, 0, stream>>>(A, b1, g1, bb1, m1, v1, B, n16);  // B = h1

  // ---- global mean pool ----
  fill_kernel<<<(g64 + G + BLK - 1) / BLK, BLK, 0, stream>>>(pooled, 0.0f, g64 + G);
  count_kernel<<<gbN, BLK, 0, stream>>>(batch, cnt, N);
  pool_sum_kernel<<<gbN64, BLK, 0, stream>>>(batch, B, pooled, n64);
  pool_div_kernel<<<(g64 + BLK - 1) / BLK, BLK, 0, stream>>>(pooled, cnt, g64);

  // ---- MLP head ----
  head_linear_kernel<<<(G * 256 + BLK - 1) / BLK, BLK, 0, stream>>>(
      pooled, hW1, hb1, hg1, hbb1, hm1, hv1, z1, G, 64, 256, 0);
  head_linear_kernel<<<(G * 128 + BLK - 1) / BLK, BLK, 0, stream>>>(
      z1, hW2, hb2, hg2, hbb2, hm2, hv2, z2, G, 256, 128, 0);
  head_linear_kernel<<<(G * 64 + BLK - 1) / BLK, BLK, 0, stream>>>(
      z2, hW3, hb3, nullptr, nullptr, nullptr, nullptr, z3, G, 128, 64, 1);
  head_linear_kernel<<<(G + BLK - 1) / BLK, BLK, 0, stream>>>(
      z3, hW4, hb4, nullptr, nullptr, nullptr, nullptr, (float*)d_out, G, 64, 1, 2);
}
